// BiLSTM_CRF_66417374266020
// MI455X (gfx1250) — matrix operand order repaired and checked by reading, not compiled, not run on hardware
//
#include <hip/hip_runtime.h>
#include <cstdint>
#include <cstddef>

// ---- problem dims ----
#define T_LEN 2048
#define B_N   64
#define I_N   50
#define H_N   100
#define NTAGS 6
#define TB    (T_LEN * B_N)   // 131072
#define G4H   400             // 4*H
#define KPADI 64              // I padded to 2x32 for wmma K
#define KPADH 128             // H padded to 4x32 for wmma K

typedef _Float16 h16;
typedef __attribute__((ext_vector_type(16))) _Float16     v16h;
typedef __attribute__((ext_vector_type(8)))  float        v8f;
typedef __attribute__((ext_vector_type(4)))  unsigned int v4u;
typedef __attribute__((ext_vector_type(4)))  int          v4i;
typedef __attribute__((ext_vector_type(8)))  int          v8i;

// ---------------- WMMA operand loaders (per ISA 7.12.2 layouts) ----------------
// A matrix 16x32 f16: lane 0-15 -> rows M=0..15 (K halfblock 0); lanes 16-31 same rows (K halfblock 8).
__device__ __forceinline__ v16h load_a_tile(const h16* base, int ld, int r0, int k0, int lane) {
    v16h a;
    int row = r0 + (lane & 15);
    int kb  = (lane >> 4) << 3;   // 0 or 8
    const h16* p = base + (size_t)row * ld + k0;
#pragma unroll
    for (int v = 0; v < 8; ++v) {
        int k = ((v < 4) ? 0 : 16) + kb + ((v & 3) << 1);
        a[2 * v]     = p[k];
        a[2 * v + 1] = p[k + 1];
    }
    return a;
}

// B matrix 32x16 f16 (KxN): lane%16 -> N, elements -> K in same order as A
__device__ __forceinline__ v16h load_b_tile(const h16* base, int ld, int k0, int n0, int lane) {
    v16h b;
    int kb = (lane >> 4) << 3;
    const h16* p = base + (size_t)k0 * ld + n0 + (lane & 15);
#pragma unroll
    for (int n = 0; n < 16; ++n) b[n] = p[(size_t)(((n < 8) ? 0 : 8) + kb + n) * ld];
    return b;
}

__device__ __forceinline__ float sigm(float x) { return 1.0f / (1.0f + __expf(-x)); }

// ---------------- TDM 1-D async copy: global -> LDS (ISA ch. 8 descriptor) ----------------
// Copies `bytes` (multiple of 8, 8B aligned both sides) to LDS byte offset lds_addr.
__device__ __forceinline__ void tdm_load_1d(unsigned int lds_addr, const void* gptr,
                                            unsigned int bytes) {
    unsigned long long ga = (unsigned long long)(uintptr_t)gptr;
    unsigned int n = bytes >> 3;  // 8-byte elements
    // group 0: count=1 [1:0] | lds_addr [63:32] | global_addr [120:64] | type=2 [127:126]
    v4u g0;
    g0[0] = 1u;
    g0[1] = lds_addr;
    g0[2] = (unsigned int)(ga & 0xFFFFFFFFull);
    g0[3] = (unsigned int)((ga >> 32) & 0x01FFFFFFull) | (2u << 30);
    // group 1: wg_mask=0 | data_size=3 (8B) [17:16] | tensor_dim0 [79:48] | tensor_dim1 [111:80]
    //          tile_dim0 [127:112] | tile_dim1 [143:128] | tile_dim2 [159:144]
    //          tensor_dim0_stride [207:160] | tensor_dim1_stride [255:208]
    v8i g1;
    g1[0] = (int)(3u << 16);                                  // data_size = 8B
    g1[1] = (int)((n & 0xFFFFu) << 16);                       // tensor_dim0[15:0]
    g1[2] = (int)(((n >> 16) & 0xFFFFu) | (1u << 16));        // tensor_dim0[31:16] | tensor_dim1=1
    g1[3] = (int)((n & 0xFFFFu) << 16);                       // tile_dim0 = n
    g1[4] = 1;                                                // tile_dim1 = 1, tile_dim2 = 0
    g1[5] = (int)n;                                           // tensor_dim0_stride
    g1[6] = 0;
    g1[7] = 0;
    v4i z4 = {0, 0, 0, 0};
#if __clang_major__ >= 23
    v8i z8 = {0, 0, 0, 0, 0, 0, 0, 0};
    __builtin_amdgcn_tensor_load_to_lds(g0, g1, z4, z4, z8, 0);
#else
    __builtin_amdgcn_tensor_load_to_lds(g0, g1, z4, z4, 0);
#endif
}

// ---------------- prep kernels (f32 -> f16, pad + transpose weights) ----------------
__global__ void k_conv_x(const float* __restrict__ x, h16* __restrict__ x16) {
    int idx = blockIdx.x * blockDim.x + threadIdx.x;
    if (idx >= TB * KPADI) return;
    int row = idx / KPADI, k = idx % KPADI;
    x16[idx] = (h16)(k < I_N ? x[(size_t)row * I_N + k] : 0.0f);
}

__global__ void k_conv_wih(const float* __restrict__ wf, const float* __restrict__ wb,
                           h16* __restrict__ wih_t) {
    int idx = blockIdx.x * blockDim.x + threadIdx.x;
    if (idx >= 2 * KPADI * G4H) return;
    int dir = idx / (KPADI * G4H);
    int r   = idx % (KPADI * G4H);
    int k = r / G4H, n = r % G4H;
    const float* w = dir ? wb : wf;
    wih_t[idx] = (h16)(k < I_N ? w[(size_t)n * I_N + k] : 0.0f);   // transposed: (K,N)
}

__global__ void k_conv_whh(const float* __restrict__ wf, const float* __restrict__ wb,
                           h16* __restrict__ whh_t) {
    int idx = blockIdx.x * blockDim.x + threadIdx.x;
    if (idx >= 2 * KPADH * G4H) return;
    int dir = idx / (KPADH * G4H);
    int r   = idx % (KPADH * G4H);
    int k = r / G4H, n = r % G4H;
    const float* w = dir ? wb : wf;
    whh_t[idx] = (h16)(k < H_N ? w[(size_t)n * H_N + k] : 0.0f);   // transposed: (K,N)
}

__global__ void k_conv_bias(const float* bif, const float* bhf,
                            const float* bib, const float* bhb, float* bias) {
    int idx = blockIdx.x * blockDim.x + threadIdx.x;
    if (idx >= 2 * G4H) return;
    int dir = idx / G4H, n = idx % G4H;
    bias[idx] = dir ? (bib[n] + bhb[n]) : (bif[n] + bhf[n]);
}

// ---------------- pregate GEMM: xg[dir] = x16 (TB x 64) * wih_t[dir] (64 x 400) + bias ----------------
__global__ void k_pregate(const h16* __restrict__ x16, const h16* __restrict__ wih_t,
                          const float* __restrict__ bias, h16* __restrict__ xg) {
    int gt   = blockIdx.x * blockDim.x + threadIdx.x;
    int wave = gt >> 5;
    int lane = gt & 31;
    const int MT = TB / 16;    // 8192
    const int NT = G4H / 16;   // 25
    if (wave >= 2 * MT * NT) return;
    int dir = wave / (MT * NT);
    int rem = wave % (MT * NT);
    int mt = rem / NT, nt = rem % NT;
    int r0 = mt * 16, n0 = nt * 16;

    const h16* W = wih_t + (size_t)dir * KPADI * G4H;
    v8f acc = {};
#pragma unroll
    for (int kk = 0; kk < 2; ++kk) {
        v16h a = load_a_tile(x16, KPADI, r0, kk * 32, lane);
        v16h b = load_b_tile(W, G4H, kk * 32, n0, lane);
        acc = __builtin_amdgcn_wmma_f32_16x16x32_f16(false, a, false, b, (short)0, acc, false, false);
    }
    int Moff = (lane >> 4) << 3;     // 0 or 8
    int N    = lane & 15;
    h16* out = xg + (size_t)dir * TB * G4H;
    float bn = bias[dir * G4H + n0 + N];
#pragma unroll
    for (int v = 0; v < 8; ++v) {
        int row = r0 + Moff + v;
        out[(size_t)row * G4H + n0 + N] = (h16)(acc[v] + bn);
    }
}

// ---------------- persistent recurrent LSTM: one block per direction ----------------
// LDS map (bytes):
#define SMEM_W   0                                    // 128x400 f16 = 102400
#define SMEM_H   (KPADH * G4H * 2)                    // 64x128  f16 = 16384
#define SMEM_G   (SMEM_H + B_N * KPADH * 2)           // 64x400  f16 = 51200
#define SMEM_C   (SMEM_G + B_N * G4H * 2)             // 64x100  f32 = 25600
#define SMEM_X0  (SMEM_C + B_N * H_N * 4)             // 64x400  f16 = 51200 (xg stage buf 0)
#define SMEM_X1  (SMEM_X0 + B_N * G4H * 2)            // 64x400  f16 = 51200 (xg stage buf 1)
#define SMEM_TOT (SMEM_X1 + B_N * G4H * 2)            // 297984 bytes < 320KB/WGP

__global__ __launch_bounds__(1024)
void k_lstm_recurrent(const h16* __restrict__ xg, const h16* __restrict__ whh_t,
                      h16* __restrict__ hcat) {
    extern __shared__ char smem[];
    h16*   sW = (h16*)(smem + SMEM_W);
    h16*   sH = (h16*)(smem + SMEM_H);
    h16*   sG = (h16*)(smem + SMEM_G);
    float* sC = (float*)(smem + SMEM_C);

    const int dir = blockIdx.x;
    const int tid = threadIdx.x;
    const int wave = tid >> 5, lane = tid & 31;

    // cooperative load of W_hh^T into LDS, zero h and c
    const h16* Wg = whh_t + (size_t)dir * KPADH * G4H;
    for (int i = tid; i < KPADH * G4H; i += 1024) sW[i] = Wg[i];
    for (int i = tid; i < B_N * KPADH; i += 1024) sH[i] = (h16)0.0f;
    for (int i = tid; i < B_N * H_N; i += 1024) sC[i] = 0.0f;

    const h16* xg_dir = xg + (size_t)dir * TB * G4H;
    const unsigned int stepBytes = B_N * G4H * 2;   // 51200

    // prologue: TDM-prefetch xg for step 0 into buffer 0 (wave 0 issues, TENSORcnt tracked)
    if (wave == 0) {
        int tt0 = dir ? (T_LEN - 1) : 0;
        tdm_load_1d((unsigned int)SMEM_X0, xg_dir + (size_t)tt0 * B_N * G4H, stepBytes);
    }
    __syncthreads();

    for (int t = 0; t < T_LEN; ++t) {
        const int tt  = dir ? (T_LEN - 1 - t) : t;
        const int cur = t & 1;
        const unsigned int curOff  = cur ? (unsigned int)SMEM_X1 : (unsigned int)SMEM_X0;
        const unsigned int nextOff = cur ? (unsigned int)SMEM_X0 : (unsigned int)SMEM_X1;

        // wave 0: ensure buffer `cur` landed, then kick off TDM for step t+1 into other buffer
        if (wave == 0) {
            __builtin_amdgcn_s_wait_tensorcnt(0);
            if (t + 1 < T_LEN) {
                int ttn = dir ? (T_LEN - 2 - t) : (t + 1);
                tdm_load_1d(nextOff, xg_dir + (size_t)ttn * B_N * G4H, stepBytes);
            }
        }
        __syncthreads();   // publish staged xg to all waves

        const h16* sXc = (const h16*)(smem + curOff);

        // G = H (64x128) * W (128x400) + xg ; 4x25 = 100 tiles over 32 waves
        for (int id = wave; id < 100; id += 32) {
            int mt = id / 25, nt = id % 25;
            int r0 = mt * 16, n0 = nt * 16;
            v8f acc = {};
#pragma unroll
            for (int kk = 0; kk < 4; ++kk) {
                v16h a = load_a_tile(sH, KPADH, r0, kk * 32, lane);
                v16h b = load_b_tile(sW, G4H, kk * 32, n0, lane);
                acc = __builtin_amdgcn_wmma_f32_16x16x32_f16(false, a, false, b, (short)0, acc,
                                                             false, false);
            }
            int Moff = (lane >> 4) << 3;
            int N    = lane & 15;
#pragma unroll
            for (int v = 0; v < 8; ++v) {
                int row = r0 + Moff + v;
                int col = n0 + N;
                sG[row * G4H + col] = (h16)(acc[v] + (float)sXc[row * G4H + col]);
            }
        }
        __syncthreads();

        // elementwise gate update: 6400 (b,h) pairs
        for (int e = tid; e < B_N * H_N; e += 1024) {
            int b = e / H_N, h = e % H_N;
            const h16* g = sG + b * G4H;
            float ig = sigm((float)g[h]);
            float fg = sigm((float)g[H_N + h]);
            float gg = tanhf((float)g[2 * H_N + h]);
            float og = sigm((float)g[3 * H_N + h]);
            float c  = fg * sC[e] + ig * gg;
            float hh = og * tanhf(c);
            sC[e] = c;
            sH[b * KPADH + h] = (h16)hh;
            hcat[((size_t)tt * B_N + b) * (2 * H_N) + dir * H_N + h] = (h16)hh;
        }
        __syncthreads();
    }
}

// ---------------- emissions: (T*B, 200) x (200, 6) + b_tag ----------------
__global__ void k_emis(const h16* __restrict__ hcat, const float* __restrict__ w_tag,
                       const float* __restrict__ b_tag, float* __restrict__ emis) {
    int idx = blockIdx.x * blockDim.x + threadIdx.x;
    if (idx >= TB * NTAGS) return;
    int tb = idx / NTAGS, j = idx % NTAGS;
    const h16*  h = hcat + (size_t)tb * (2 * H_N);
    const float* w = w_tag + (size_t)j * (2 * H_N);
    float s = b_tag[j];
#pragma unroll 4
    for (int k = 0; k < 2 * H_N; ++k) s += (float)h[k] * w[k];
    emis[idx] = s;
}

// ---------------- Viterbi forward + backtrace (one block, 64 chains x 6 tags) ----------------
__global__ __launch_bounds__(384)
void k_viterbi(const float* __restrict__ emis, const float* __restrict__ start_t,
               const float* __restrict__ end_t, const float* __restrict__ trans,
               unsigned char* __restrict__ bp, int* __restrict__ tags) {
    __shared__ float sS[B_N * NTAGS];
    __shared__ float sNw[B_N * NTAGS];
    __shared__ float sT[NTAGS * NTAGS];
    __shared__ float sE[NTAGS];
    const int tid = threadIdx.x;            // 384 threads
    const int b = tid / NTAGS, j = tid % NTAGS;

    if (tid < NTAGS * NTAGS) sT[tid] = trans[tid];
    if (tid < NTAGS) sE[tid] = end_t[tid];
    sS[tid] = start_t[j] + emis[(size_t)b * NTAGS + j];   // t = 0
    __syncthreads();

    for (int t = 1; t < T_LEN; ++t) {
        float e = emis[((size_t)t * B_N + b) * NTAGS + j];
        float best = -3.0e38f;
        int bi = 0;
#pragma unroll
        for (int i = 0; i < NTAGS; ++i) {
            float v = sS[b * NTAGS + i] + sT[i * NTAGS + j];
            if (v > best) { best = v; bi = i; }
        }
        sNw[tid] = best + e;
        bp[((size_t)t * B_N + b) * NTAGS + j] = (unsigned char)bi;
        __syncthreads();
        sS[tid] = sNw[tid];
        __syncthreads();
    }

    // backtrace: one thread per batch element
    if (tid < B_N) {
        int bb = tid;
        float best = -3.0e38f;
        int cur = 0;
#pragma unroll
        for (int jj = 0; jj < NTAGS; ++jj) {
            float v = sS[bb * NTAGS + jj] + sE[jj];
            if (v > best) { best = v; cur = jj; }
        }
        tags[(size_t)(T_LEN - 1) * B_N + bb] = cur;
        for (int t = T_LEN - 2; t >= 0; --t) {
            cur = bp[((size_t)(t + 1) * B_N + bb) * NTAGS + cur];
            tags[(size_t)t * B_N + bb] = cur;
        }
    }
}

// ---------------- one-hot output ----------------
__global__ void k_onehot(const int* __restrict__ tags, float* __restrict__ out) {
    int idx = blockIdx.x * blockDim.x + threadIdx.x;
    if (idx >= TB) return;
    int tg = tags[idx];
#pragma unroll
    for (int j = 0; j < NTAGS; ++j) out[(size_t)idx * NTAGS + j] = (j == tg) ? 1.0f : 0.0f;
}

// ---------------- host launch ----------------
extern "C" void kernel_launch(void* const* d_in, const int* in_sizes, int n_in,
                              void* d_out, int out_size, void* d_ws, size_t ws_size,
                              hipStream_t stream) {
    const float* x      = (const float*)d_in[0];
    const float* w_ih_f = (const float*)d_in[1];
    const float* w_hh_f = (const float*)d_in[2];
    const float* b_ih_f = (const float*)d_in[3];
    const float* b_hh_f = (const float*)d_in[4];
    const float* w_ih_b = (const float*)d_in[5];
    const float* w_hh_b = (const float*)d_in[6];
    const float* b_ih_b = (const float*)d_in[7];
    const float* b_hh_b = (const float*)d_in[8];
    const float* w_tag  = (const float*)d_in[9];
    const float* b_tag  = (const float*)d_in[10];
    const float* start_t = (const float*)d_in[11];
    const float* end_t   = (const float*)d_in[12];
    const float* trans   = (const float*)d_in[13];
    float* out = (float*)d_out;

    // workspace carve-up (256B aligned segments)
    char* ws = (char*)d_ws;
    size_t off = 0;
    auto carve = [&](size_t bytes) -> char* {
        char* p = ws + off;
        off = (off + bytes + 255) & ~(size_t)255;
        return p;
    };
    h16*  x16    = (h16*)carve((size_t)TB * KPADI * 2);          // 16 MB
    h16*  wih_t  = (h16*)carve((size_t)2 * KPADI * G4H * 2);     // 100 KB
    h16*  whh_t  = (h16*)carve((size_t)2 * KPADH * G4H * 2);     // 200 KB
    float* bias  = (float*)carve((size_t)2 * G4H * 4);           // 3.2 KB
    h16*  xg     = (h16*)carve((size_t)2 * TB * G4H * 2);        // 210 MB
    h16*  hcat   = (h16*)carve((size_t)TB * 2 * H_N * 2);        // 52 MB
    float* emis  = (float*)carve((size_t)TB * NTAGS * 4);        // 3 MB
    unsigned char* bp = (unsigned char*)carve((size_t)T_LEN * B_N * NTAGS); // 0.75 MB
    int*  tags   = (int*)carve((size_t)TB * 4);                  // 0.5 MB
    (void)ws_size; (void)in_sizes; (void)n_in; (void)out_size;

    // 1) convert/pad/transpose inputs to f16
    k_conv_x<<<(TB * KPADI + 255) / 256, 256, 0, stream>>>(x, x16);
    k_conv_wih<<<(2 * KPADI * G4H + 255) / 256, 256, 0, stream>>>(w_ih_f, w_ih_b, wih_t);
    k_conv_whh<<<(2 * KPADH * G4H + 255) / 256, 256, 0, stream>>>(w_hh_f, w_hh_b, whh_t);
    k_conv_bias<<<(2 * G4H + 255) / 256, 256, 0, stream>>>(b_ih_f, b_hh_f, b_ih_b, b_hh_b, bias);

    // 2) big WMMA pregate GEMM: one wave per 16x16 tile
    {
        long long waves = 2LL * (TB / 16) * (G4H / 16);  // 409600
        long long thr = waves * 32;
        k_pregate<<<(unsigned)((thr + 255) / 256), 256, 0, stream>>>(x16, wih_t, bias, xg);
    }

    // 3) persistent recurrent scan, one WGP-resident block per direction, TDM double-buffered xg
    k_lstm_recurrent<<<2, 1024, SMEM_TOT, stream>>>(xg, whh_t, hcat);

    // 4) emissions
    k_emis<<<(TB * NTAGS + 255) / 256, 256, 0, stream>>>(hcat, w_tag, b_tag, emis);

    // 5) Viterbi decode
    k_viterbi<<<1, 384, 0, stream>>>(emis, start_t, end_t, trans, bp, tags);

    // 6) one-hot output
    k_onehot<<<(TB + 255) / 256, 256, 0, stream>>>(tags, out);
}
